// GTConv_59390807769608
// MI455X (gfx1250) — compile-verified, matrix operand order checked by reading, not run on hardware
//
#include <hip/hip_runtime.h>
#include <hip/hip_bf16.h>

typedef __attribute__((ext_vector_type(16))) _Float16 v16h;
typedef __attribute__((ext_vector_type(8)))  float    v8f;

#define N_NODES 100000
#define N_EDGES 800000
#define DMODEL  128
#define NHEADS  8
#define DHEAD   16
#define QKV_W   384   // 3*DMODEL

// ---------- helpers ----------

// order-preserving encoding of float for unsigned atomicMax
__device__ __forceinline__ unsigned int enc_f32(float f) {
    unsigned int u = __float_as_uint(f);
    return (u & 0x80000000u) ? ~u : (u | 0x80000000u);
}
__device__ __forceinline__ float dec_f32(unsigned int e) {
    unsigned int u = (e & 0x80000000u) ? (e & 0x7FFFFFFFu) : ~e;
    return __uint_as_float(u);
}
#define ENC_NEG_INF 0x007FFFFFu   // enc(-inf)

// load 16 contiguous fp32, convert to f16 operand vector (global_load_b128 x4 + v_cvt_pk_f16_f32)
__device__ __forceinline__ v16h load16_cvt(const float* __restrict__ p) {
    const float4* q = reinterpret_cast<const float4*>(p);
    float4 t0 = q[0], t1 = q[1], t2 = q[2], t3 = q[3];
    v16h r;
    r[0]=(_Float16)t0.x; r[1]=(_Float16)t0.y; r[2]=(_Float16)t0.z; r[3]=(_Float16)t0.w;
    r[4]=(_Float16)t1.x; r[5]=(_Float16)t1.y; r[6]=(_Float16)t1.z; r[7]=(_Float16)t1.w;
    r[8]=(_Float16)t2.x; r[9]=(_Float16)t2.y; r[10]=(_Float16)t2.z; r[11]=(_Float16)t2.w;
    r[12]=(_Float16)t3.x; r[13]=(_Float16)t3.y; r[14]=(_Float16)t3.z; r[15]=(_Float16)t3.w;
    return r;
}

#define WMMA_F16(A_, B_, C_) \
    __builtin_amdgcn_wmma_f32_16x16x32_f16(false, (A_), false, (B_), (short)0, (C_), false, false)

// ---------- kernel 0: workspace init ----------
__global__ __launch_bounds__(256) void k_init(float* __restrict__ accum,
                                              float* __restrict__ denom,
                                              unsigned int* __restrict__ m_enc) {
    int idx = blockIdx.x * blockDim.x + threadIdx.x;
    if (idx < N_NODES * DMODEL) accum[idx] = 0.0f;
    if (idx < N_NODES * NHEADS) { denom[idx] = 0.0f; m_enc[idx] = ENC_NEG_INF; }
}

// ---------- WMMA GEMM: C[M, tilesN4*64] = A[M,128] * W[tilesN4*64,128]^T + bias ----------
// one wave per 16x64 C strip: 4 N-tiles share one A operand (4x less A traffic),
// K=128 in 4 steps of v_wmma_f32_16x16x32_f16 -> 16 WMMAs per wave.
__global__ __launch_bounds__(256) void k_gemm_wmma_bias(const float* __restrict__ A,
                                                        const float* __restrict__ W,
                                                        const float* __restrict__ bias,
                                                        float* __restrict__ C,
                                                        int tilesM, int tilesN4) {
    const int lane = threadIdx.x & 31;
    const int wave = blockIdx.x * (blockDim.x >> 5) + (threadIdx.x >> 5);
    if (wave >= tilesM * tilesN4) return;             // wave-uniform guard, EXEC stays all-1
    const int tm  = wave / tilesN4;
    const int tn0 = (wave % tilesN4) * 4;             // first of 4 consecutive 16-col tiles
    const int ldc = tilesN4 * 64;

    const int mrow  = tm * 16 + (lane & 15);          // A row this lane supplies
    const int nbase = tn0 * 16 + (lane & 15);         // W row (= C column) for tile 0
    const int khalf = (lane >> 4) << 4;               // lanes 16..31 hold K+16..K+31

    v8f c0 = {}, c1 = {}, c2 = {}, c3 = {};
#pragma unroll
    for (int kk = 0; kk < DMODEL; kk += 32) {
        v16h a  = load16_cvt(A + (size_t)mrow * DMODEL + kk + khalf);
        v16h b0 = load16_cvt(W + (size_t)(nbase +  0) * DMODEL + kk + khalf);
        v16h b1 = load16_cvt(W + (size_t)(nbase + 16) * DMODEL + kk + khalf);
        v16h b2 = load16_cvt(W + (size_t)(nbase + 32) * DMODEL + kk + khalf);
        v16h b3 = load16_cvt(W + (size_t)(nbase + 48) * DMODEL + kk + khalf);
        c0 = WMMA_F16(a, b0, c0);
        c1 = WMMA_F16(a, b1, c1);
        c2 = WMMA_F16(a, b2, c2);
        c3 = WMMA_F16(a, b3, c3);
    }

    const int col   = lane & 15;
    const int rbase = (lane >> 4) * 8;                // VGPR r holds M = r (+8 for hi lanes)
    const float bv0 = bias[tn0 * 16 +  0 + col];
    const float bv1 = bias[tn0 * 16 + 16 + col];
    const float bv2 = bias[tn0 * 16 + 32 + col];
    const float bv3 = bias[tn0 * 16 + 48 + col];
    float* rowp = C + (size_t)(tm * 16 + rbase) * ldc + tn0 * 16 + col;
#pragma unroll
    for (int r = 0; r < 8; ++r) {
        rowp[(size_t)r * ldc +  0] = c0[r] + bv0;
        rowp[(size_t)r * ldc + 16] = c1[r] + bv1;
        rowp[(size_t)r * ldc + 32] = c2[r] + bv2;
        rowp[(size_t)r * ldc + 48] = c3[r] + bv3;
    }
}

// ---------- kernel 2: per-(edge,head) score + segmented atomic max ----------
__global__ __launch_bounds__(256) void k_edge_score(const float* __restrict__ qkv,
                                                    const int* __restrict__ src,
                                                    const int* __restrict__ dst,
                                                    float* __restrict__ scores,
                                                    unsigned int* __restrict__ m_enc) {
    int idx = blockIdx.x * blockDim.x + threadIdx.x;
    if (idx >= N_EDGES * NHEADS) return;
    const int e = idx >> 3, h = idx & 7;
    const int s = src[e], d = dst[e];
    const float4* kp = reinterpret_cast<const float4*>(qkv + (size_t)s * QKV_W + h * 48 + DHEAD);
    const float4* qp = reinterpret_cast<const float4*>(qkv + (size_t)d * QKV_W + h * 48);
    float acc = 0.0f;
#pragma unroll
    for (int i = 0; i < 4; ++i) {
        float4 kv = kp[i], qv = qp[i];
        acc = fmaf(kv.x, qv.x, acc); acc = fmaf(kv.y, qv.y, acc);
        acc = fmaf(kv.z, qv.z, acc); acc = fmaf(kv.w, qv.w, acc);
    }
    const float sc = acc * 0.25f;                     // DHEAD^-0.5
    scores[idx] = sc;
    atomicMax(&m_enc[(size_t)d * NHEADS + h], enc_f32(sc));
}

// ---------- kernel 3: p = exp(score - m), denom += p ----------
__global__ __launch_bounds__(256) void k_edge_exp(const int* __restrict__ dst,
                                                  float* __restrict__ scores,
                                                  const unsigned int* __restrict__ m_enc,
                                                  float* __restrict__ denom) {
    int idx = blockIdx.x * blockDim.x + threadIdx.x;
    if (idx >= N_EDGES * NHEADS) return;
    const int e = idx >> 3, h = idx & 7;
    const int d = dst[e];
    const float m = dec_f32(m_enc[(size_t)d * NHEADS + h]);
    const float p = __expf(scores[idx] - m);
    scores[idx] = p;                                  // overwrite with p
    atomicAdd(&denom[(size_t)d * NHEADS + h], p);
}

// ---------- kernel 4: accum[dst] += v[src] * p/denom ----------
__global__ __launch_bounds__(256) void k_edge_scatter(const float* __restrict__ qkv,
                                                      const int* __restrict__ src,
                                                      const int* __restrict__ dst,
                                                      const float* __restrict__ scores,
                                                      const float* __restrict__ denom,
                                                      float* __restrict__ accum) {
    int idx = blockIdx.x * blockDim.x + threadIdx.x;
    if (idx >= N_EDGES * NHEADS) return;
    const int e = idx >> 3, h = idx & 7;
    const int s = src[e], d = dst[e];
    const float prob = scores[idx] / denom[(size_t)d * NHEADS + h];
    const float4* vp = reinterpret_cast<const float4*>(qkv + (size_t)s * QKV_W + h * 48 + 2 * DHEAD);
    float* op = accum + (size_t)d * DMODEL + h * DHEAD;
#pragma unroll
    for (int i = 0; i < 4; ++i) {
        float4 v = vp[i];
        atomicAdd(op + 4 * i + 0, v.x * prob);
        atomicAdd(op + 4 * i + 1, v.y * prob);
        atomicAdd(op + 4 * i + 2, v.z * prob);
        atomicAdd(op + 4 * i + 3, v.w * prob);
    }
}

// ---------- launch ----------
extern "C" void kernel_launch(void* const* d_in, const int* in_sizes, int n_in,
                              void* d_out, int out_size, void* d_ws, size_t ws_size,
                              hipStream_t stream) {
    const float* x     = (const float*)d_in[0];   // [N, 128]
    const float* W_qkv = (const float*)d_in[1];   // [384, 128]
    const float* b_qkv = (const float*)d_in[2];   // [384]
    const float* W_out = (const float*)d_in[3];   // [128, 128]
    const float* b_out = (const float*)d_in[4];   // [128]
    const int*   src   = (const int*)d_in[5];     // [E]
    const int*   dst   = (const int*)d_in[6];     // [E]
    float* out = (float*)d_out;                   // [N, 128]

    // workspace layout (all sizes 256B-aligned)
    char* ws = (char*)d_ws;
    const size_t QKV_BYTES = (size_t)N_NODES * QKV_W * sizeof(float);          // 153.6 MB
    const size_t SC_BYTES  = (size_t)N_EDGES * NHEADS * sizeof(float);         //  25.6 MB
    const size_t MH_BYTES  = (size_t)N_NODES * NHEADS * sizeof(unsigned int);  //   3.2 MB
    const size_t DEN_BYTES = (size_t)N_NODES * NHEADS * sizeof(float);         //   3.2 MB
    float*        qkv    = (float*)(ws);
    float*        scores = (float*)(ws + QKV_BYTES);
    unsigned int* m_enc  = (unsigned int*)(ws + QKV_BYTES + SC_BYTES);
    float*        denom  = (float*)(ws + QKV_BYTES + SC_BYTES + MH_BYTES);
    float*        accum  = (float*)(ws + QKV_BYTES + SC_BYTES + MH_BYTES + DEN_BYTES);

    // 0) init accum/denom/max
    {
        int n = N_NODES * DMODEL;
        k_init<<<(n + 255) / 256, 256, 0, stream>>>(accum, denom, m_enc);
    }
    // 1) QKV GEMM: 6250 x 6 strips of 16x64, 8 waves/block
    {
        int tilesM = N_NODES / 16, tilesN4 = QKV_W / 64;    // 6250 x 6
        int waves  = tilesM * tilesN4;                       // 37500
        k_gemm_wmma_bias<<<(waves + 7) / 8, 256, 0, stream>>>(x, W_qkv, b_qkv, qkv, tilesM, tilesN4);
    }
    // 2) scores + segment max
    {
        int n = N_EDGES * NHEADS;
        k_edge_score<<<(n + 255) / 256, 256, 0, stream>>>(qkv, src, dst, scores, m_enc);
    }
    // 3) exp + segment sum
    {
        int n = N_EDGES * NHEADS;
        k_edge_exp<<<(n + 255) / 256, 256, 0, stream>>>(dst, scores, m_enc, denom);
    }
    // 4) weighted scatter of V
    {
        int n = N_EDGES * NHEADS;
        k_edge_scatter<<<(n + 255) / 256, 256, 0, stream>>>(qkv, src, dst, scores, denom, accum);
    }
    // 5) output GEMM: 6250 x 2 strips of 16x64
    {
        int tilesM = N_NODES / 16, tilesN4 = DMODEL / 64;   // 6250 x 2
        int waves  = tilesM * tilesN4;                       // 12500
        k_gemm_wmma_bias<<<(waves + 7) / 8, 256, 0, stream>>>(accum, W_out, b_out, out, tilesM, tilesN4);
    }
}
